// Coref_45174466019998
// MI455X (gfx1250) — compile-verified, matrix operand order checked by reading, not run on hardware
//
#include <hip/hip_runtime.h>
#include <hip/hip_bf16.h>

// CDNA5 (gfx1250) wave32 WMMA types
typedef __attribute__((ext_vector_type(16))) _Float16 v16h;
typedef __attribute__((ext_vector_type(8)))  _Float16 v8h;
typedef __attribute__((ext_vector_type(8)))  float    v8f;

#define T_TOK   20000
#define HID     150      // scorer/attn hidden dim
#define HPAD    160      // HID padded to multiple of 32
#define NTILES  10       // ceil(150/16)
#define SPB     64       // spans per block in fused span kernel

// ---------------------------------------------------------------------------
// Pack an f32 [M x K] activation into f16 [M x Kpad], zero-padded K tail.
// ---------------------------------------------------------------------------
__global__ void pack_a_f16(const float* __restrict__ src, _Float16* __restrict__ dst,
                           int K, int Kpad, long total)
{
    const long i = (long)blockIdx.x * blockDim.x + threadIdx.x;
    if (i >= total) return;
    const int  k   = (int)(i % Kpad);
    const long row = i / Kpad;
    dst[i] = (_Float16)((k < K) ? src[row * K + k] : 0.0f);
}

// ---------------------------------------------------------------------------
// Pack f32 weight B [K x N] (ldb) into WMMA-B fragment order, f16, zero-padded:
//   dst[((nt*KT + kt)*32 + lane)*16 + e],  kk = kt*32 + (lane/16)*8 + (e<8?e:e+8),
//   col = nt*16 + lane%16.  One aligned v16h per (wave-lane, nt, kt) at use site.
// ---------------------------------------------------------------------------
__global__ void pack_b_frag(const float* __restrict__ B, _Float16* __restrict__ dst,
                            int K, int N, int ldb, int KT, int NT)
{
    const int i = blockIdx.x * blockDim.x + threadIdx.x;
    if (i >= NT * KT * 512) return;
    const int e    = i & 15;
    const int lane = (i >> 4) & 31;
    const int rem  = i >> 9;            // nt*KT + kt
    const int kt   = rem % KT;
    const int nt   = rem / KT;
    const int kk   = kt * 32 + ((lane >> 4) << 3) + (e < 8 ? e : e + 8);
    const int col  = nt * 16 + (lane & 15);
    dst[i] = (_Float16)((kk < K && col < N) ? B[kk * ldb + col] : 0.0f);
}

// ---------------------------------------------------------------------------
// Branchless WMMA GEMM: act(A[M x KT*32](f16,padded) @ B(frag-packed) + bias).
// 8 waves/block = 128 rows; each wave owns two adjacent 16x16 C tiles
// (A-fragment reuse). Fully unrolled K loop (KT is a template constant).
// Output: f32 [M x N] and/or f16 [M x Npad] (zero-padded) for chaining.
// ---------------------------------------------------------------------------
template <int KT>
__global__ void wmma_gemm2(const _Float16* __restrict__ A,
                           const _Float16* __restrict__ Bp,
                           const float*    __restrict__ bias,
                           float*          __restrict__ Cf32,
                           _Float16*       __restrict__ Cf16,
                           int M, int N, int Npad, int relu)
{
    const int lane = threadIdx.x & 31;
    const int wave = threadIdx.x >> 5;
    const int row0 = (blockIdx.y * 8 + wave) * 16;
    if (row0 >= M) return;                    // wave-uniform (M % 16 == 0)
    const int nt0  = blockIdx.x * 2;          // two adjacent N tiles
    const int Kpad = KT * 32;

    const _Float16* arow = A + (size_t)(row0 + (lane & 15)) * Kpad + ((lane >> 4) << 3);
    const _Float16* bp0  = Bp + ((size_t)(nt0    ) * KT * 32 + lane) * 16;
    const _Float16* bp1  = Bp + ((size_t)(nt0 + 1) * KT * 32 + lane) * 16;

    v8f acc0 = {}, acc1 = {};
#pragma unroll
    for (int kt = 0; kt < KT; ++kt) {
        const v8h alo = *(const v8h*)(arow + kt * 32);
        const v8h ahi = *(const v8h*)(arow + kt * 32 + 16);
        v16h a;
#pragma unroll
        for (int e = 0; e < 8; ++e) { a[e] = alo[e]; a[e + 8] = ahi[e]; }
        const v16h b0 = *(const v16h*)(bp0 + kt * 512);
        const v16h b1 = *(const v16h*)(bp1 + kt * 512);
        acc0 = __builtin_amdgcn_wmma_f32_16x16x32_f16(false, a, false, b0,
                                                      (short)0, acc0, false, false);
        acc1 = __builtin_amdgcn_wmma_f32_16x16x32_f16(false, a, false, b1,
                                                      (short)0, acc1, false, false);
    }

    // C/D layout: VGPR v -> row v (+8 for lanes 16..31), col = lane%16
    const int colA  = nt0 * 16 + (lane & 15);
    const int colB  = colA + 16;
    const int rbase = row0 + (lane < 16 ? 0 : 8);
    float bia0 = 0.0f, bia1 = 0.0f;
    if (bias) {
        bia0 = (colA < N) ? bias[colA] : 0.0f;
        bia1 = (colB < N) ? bias[colB] : 0.0f;
    }
#pragma unroll
    for (int v = 0; v < 8; ++v) {
        const int r = rbase + v;                       // always < M (M % 16 == 0)
        float v0 = acc0[v] + bia0;
        float v1 = acc1[v] + bia1;
        if (relu) { v0 = fmaxf(v0, 0.0f); v1 = fmaxf(v1, 0.0f); }
        if (Cf32) {
            if (colA < N) Cf32[(size_t)r * N + colA] = v0;
            if (colB < N) Cf32[(size_t)r * N + colB] = v1;
        }
        if (Cf16) {
            Cf16[(size_t)r * Npad + colA] = (_Float16)((colA < N) ? v0 : 0.0f);
            Cf16[(size_t)r * Npad + colB] = (_Float16)((colB < N) ? v1 : 0.0f);
        }
    }
}

// ---------------------------------------------------------------------------
// attn layer 3 (matvec): logits[t] = H2[t,:] . W3 + b3
// ---------------------------------------------------------------------------
__global__ void attn_logits_kernel(const float* __restrict__ H2,
                                   const float* __restrict__ W3,
                                   const float* __restrict__ b3,
                                   float* __restrict__ out, int T)
{
    const int t = blockIdx.x * blockDim.x + threadIdx.x;
    if (t >= T) return;
    float acc = b3[0];
    const float* row = H2 + (size_t)t * HID;
    for (int k = 0; k < HID; ++k) acc += row[k] * W3[k];
    out[t] = acc;
}

// ---------------------------------------------------------------------------
// Fused span scorer: block = 128 threads (4 waves) handles 64 spans of one
// width n = blockIdx.y+1. Softmax-pool -> h1 (LDS f16) -> WMMA layer-2 with
// pre-packed sc_W2 fragments -> h2 (LDS f16) -> dot with sc_W3.
// h1/h2 never touch HBM.
// ---------------------------------------------------------------------------
__global__ void span_score_kernel(const float* __restrict__ PS,
                                  const float* __restrict__ PE,
                                  const float* __restrict__ PM,
                                  const float* __restrict__ logits,
                                  const float* __restrict__ b1,
                                  const _Float16* __restrict__ W2p,  // frag-packed
                                  const float* __restrict__ b2,
                                  const float* __restrict__ W3,
                                  const float* __restrict__ b3,
                                  float* __restrict__ out, int T)
{
    __shared__ alignas(32) _Float16 sH1[SPB][HPAD];   // 20 KB
    __shared__ alignas(32) _Float16 sH2[SPB][HPAD];   // 20 KB
    __shared__ float sW[SPB][10];                     // 2.5 KB

    const int n    = blockIdx.y + 1;                  // span width 1..10
    const int S    = T - n + 1;                       // #spans of width n
    const int s0   = blockIdx.x * SPB;
    const int base = (n - 1) * (T + 1) - (n - 1) * n / 2;
    const int tid  = threadIdx.x;

    // --- 1) per-span softmax over the window of attn logits ---
    for (int sl = tid; sl < SPB; sl += blockDim.x) {
        const int s = s0 + sl;
        if (s < S) {
            float a[10], m = -1e30f;
            for (int j = 0; j < n; ++j) { a[j] = logits[s + j]; m = fmaxf(m, a[j]); }
            float z = 0.0f;
            for (int j = 0; j < n; ++j) { a[j] = __expf(a[j] - m); z += a[j]; }
            const float inv = 1.0f / z;
            for (int j = 0; j < n; ++j) sW[sl][j] = a[j] * inv;
        } else {
            for (int j = 0; j < n; ++j) sW[sl][j] = 0.0f;
        }
    }
    __syncthreads();

    // --- 2) h1 = relu(PS[s] + PE[e] + sum_j w_j*PM[s+j] + b1), f16 in LDS ---
    for (int e = tid; e < SPB * HPAD; e += blockDim.x) {
        const int sl = e / HPAD, k = e % HPAD;
        const int s  = s0 + sl;
        float h = 0.0f;
        if (s < S && k < HID) {
            h = PS[(size_t)s * HID + k] + PE[(size_t)(s + n - 1) * HID + k] + b1[k];
            for (int j = 0; j < n; ++j)
                h += sW[sl][j] * PM[(size_t)(s + j) * HID + k];
            h = fmaxf(h, 0.0f);
        }
        sH1[sl][k] = (_Float16)h;
    }
    __syncthreads();

    // --- 3) h2 = relu(h1 @ W2 + b2) via WMMA; wave w owns rows 16w..16w+15 ---
    const int lane = threadIdx.x & 31;
    const int wave = threadIdx.x >> 5;
    const int rr   = lane & 15;
    const int kb   = (lane >> 4) << 3;
    const _Float16* __restrict__ h1row = &sH1[wave * 16 + rr][kb];
#pragma unroll
    for (int nt = 0; nt < NTILES; ++nt) {
        v8f acc = {};
        const int col = nt * 16 + rr;
#pragma unroll
        for (int kt = 0; kt < 5; ++kt) {              // Kpad 160 / 32
            const v8h alo = *(const v8h*)(h1row + kt * 32);
            const v8h ahi = *(const v8h*)(h1row + kt * 32 + 16);
            v16h a;
#pragma unroll
            for (int e = 0; e < 8; ++e) { a[e] = alo[e]; a[e + 8] = ahi[e]; }
            const v16h b = *(const v16h*)(W2p + ((size_t)(nt * 5 + kt) * 32 + lane) * 16);
            acc = __builtin_amdgcn_wmma_f32_16x16x32_f16(false, a, false, b,
                                                         (short)0, acc, false, false);
        }
        const float bb = (col < HID) ? b2[col] : 0.0f;
#pragma unroll
        for (int v = 0; v < 8; ++v) {
            const int r = wave * 16 + v + (lane < 16 ? 0 : 8);
            sH2[r][col] = (_Float16)fmaxf(acc[v] + bb, 0.0f);
        }
    }
    __syncthreads();

    // --- 4) score = h2 . W3 + b3 ---
    for (int sl = tid; sl < SPB; sl += blockDim.x) {
        const int s = s0 + sl;
        if (s < S) {
            float acc = b3[0];
            for (int k = 0; k < HID; ++k)
                acc += (float)sH2[sl][k] * W3[k];
            out[base + s] = acc;
        }
    }
}

// ---------------------------------------------------------------------------
extern "C" void kernel_launch(void* const* d_in, const int* in_sizes, int n_in,
                              void* d_out, int out_size, void* d_ws, size_t ws_size,
                              hipStream_t stream)
{
    (void)in_sizes; (void)n_in; (void)out_size; (void)ws_size;

    const float* embeds  = (const float*)d_in[0];   // [20000,300]
    const float* states  = (const float*)d_in[1];   // [20000,400]
    const float* attn_W1 = (const float*)d_in[2];   // [400,150]
    const float* attn_b1 = (const float*)d_in[3];
    const float* attn_W2 = (const float*)d_in[4];   // [150,150]
    const float* attn_b2 = (const float*)d_in[5];
    const float* attn_W3 = (const float*)d_in[6];   // [150]
    const float* attn_b3 = (const float*)d_in[7];
    const float* sc_W1   = (const float*)d_in[8];   // [1100,150] = [W1a;W1b;W1c]
    const float* sc_b1   = (const float*)d_in[9];
    const float* sc_W2   = (const float*)d_in[10];  // [150,150]
    const float* sc_b2   = (const float*)d_in[11];
    const float* sc_W3   = (const float*)d_in[12];  // [150]
    const float* sc_b3   = (const float*)d_in[13];
    float* out = (float*)d_out;                     // [199955]

    // ---- workspace layout (256B-aligned; stream-ordered reuse) ----
    char* ws = (char*)d_ws;
    size_t off = 0;
    auto take = [&](size_t bytes) { char* p = ws + off;
                                    off = (off + bytes + 255) & ~(size_t)255; return p; };
    _Float16* statesH = (_Float16*)take((size_t)T_TOK * 416 * 2); // f16 states, Kpad 416
    _Float16* embedsH = (_Float16*)take((size_t)T_TOK * 320 * 2); // f16 embeds, Kpad 320
    _Float16* H1aH    = (_Float16*)take((size_t)T_TOK * HPAD * 2);// attn layer-1 act (padded)
    float*    bufPS   = (float*)take((size_t)T_TOK * HID * 4);    // H2a first, then PS
    float*    bufPE   = (float*)take((size_t)T_TOK * HID * 4);    // PE
    float*    logits  = (float*)take((size_t)T_TOK * 4);
    _Float16* pW1     = (_Float16*)take((size_t)NTILES * 13 * 512 * 2);
    _Float16* pW2     = (_Float16*)take((size_t)NTILES *  5 * 512 * 2);
    _Float16* pSa     = (_Float16*)take((size_t)NTILES * 13 * 512 * 2);
    _Float16* pSb     = (_Float16*)take((size_t)NTILES * 13 * 512 * 2);
    _Float16* pSc     = (_Float16*)take((size_t)NTILES * 10 * 512 * 2);
    _Float16* pSW2    = (_Float16*)take((size_t)NTILES *  5 * 512 * 2);
    float*    bufPM   = (float*)statesH;  // statesH dead before PM is written (12MB<16.6MB)

    // ---- prep: pad activations to f16, pack weights into fragment order ----
    {
        const long tS = (long)T_TOK * 416, tE = (long)T_TOK * 320;
        pack_a_f16<<<dim3((unsigned)((tS + 255) / 256)), dim3(256), 0, stream>>>(
            states, statesH, 400, 416, tS);
        pack_a_f16<<<dim3((unsigned)((tE + 255) / 256)), dim3(256), 0, stream>>>(
            embeds, embedsH, 300, 320, tE);
        auto packb = [&](const float* B, _Float16* dst, int K, int KT) {
            const int tot = NTILES * KT * 512;
            pack_b_frag<<<dim3((tot + 255) / 256), dim3(256), 0, stream>>>(
                B, dst, K, HID, HID, KT, NTILES);
        };
        packb(attn_W1,             pW1,  400, 13);
        packb(attn_W2,             pW2,  150,  5);
        packb(sc_W1,               pSa,  400, 13);
        packb(sc_W1 + 400 * HID,   pSb,  400, 13);
        packb(sc_W1 + 800 * HID,   pSc,  300, 10);
        packb(sc_W2,               pSW2, 150,  5);
    }

    dim3 blk(256);
    dim3 grd(HPAD / 32, (T_TOK + 127) / 128);         // (5, 157)

    // attn MLP: layer1 -> f16 padded act; layer2 -> f32; layer3 matvec
    wmma_gemm2<13><<<grd, blk, 0, stream>>>(statesH, pW1, attn_b1,
                                            nullptr, H1aH, T_TOK, HID, HPAD, 1);
    wmma_gemm2< 5><<<grd, blk, 0, stream>>>(H1aH, pW2, attn_b2,
                                            bufPS /*H2a*/, nullptr, T_TOK, HID, HPAD, 1);
    attn_logits_kernel<<<dim3((T_TOK + 255) / 256), dim3(256), 0, stream>>>(
        bufPS, attn_W3, attn_b3, logits, T_TOK);

    // layer-1 decomposition precompute: PS, PE, PM (f32, no bias/relu)
    wmma_gemm2<13><<<grd, blk, 0, stream>>>(statesH, pSa, nullptr,
                                            bufPS, nullptr, T_TOK, HID, HPAD, 0); // PS
    wmma_gemm2<13><<<grd, blk, 0, stream>>>(statesH, pSb, nullptr,
                                            bufPE, nullptr, T_TOK, HID, HPAD, 0); // PE
    wmma_gemm2<10><<<grd, blk, 0, stream>>>(embedsH, pSc, nullptr,
                                            bufPM, nullptr, T_TOK, HID, HPAD, 0); // PM

    // fused span scoring
    dim3 sgrd((T_TOK + SPB - 1) / SPB, 10);           // (313, 10)
    span_score_kernel<<<sgrd, dim3(128), 0, stream>>>(
        bufPS, bufPE, bufPM, logits, sc_b1, pSW2, sc_b2, sc_W3, sc_b3, out, T_TOK);
}
